// LPKTPlusV2_11493332484346
// MI455X (gfx1250) — matrix operand order repaired and checked by reading, not compiled, not run on hardware
//
#include <hip/hip_runtime.h>
#include <hip/hip_bf16.h>

// ---------------------------------------------------------------------------
// LPKT+ on MI455X (gfx1250, wave32, WMMA).
// Precompute: fold all non-recurrent linear algebra into small gather tables
// (L2 resident, W2/W3 twins packed as float2). Recurrent kernel: 16 persistent
// WGs x 8 waves; per step, 5 GEMMs [16x256x256] via v_wmma_f32_16x16x32_bf16
// with gather-initialized C accumulators; state in LDS pre-swizzled into
// A-operand register order (no repack moves).
// ---------------------------------------------------------------------------

#define Bn   256
#define Sn   500
#define TSn  (Sn - 1)      // 499 recurrent steps
#define DKn  256
#define NQ1  5001
#define NT1  101           // NIT+1 == NUT+1 == 101

typedef __attribute__((ext_vector_type(16))) __bf16 bf16x16;
typedef __attribute__((ext_vector_type(8)))  float  f32x8;

__device__ __forceinline__ float sigm(float x) { return 1.0f / (1.0f + __expf(-x)); }
__device__ __forceinline__ float tanh_fast(float x) { return 1.0f - 2.0f / (__expf(2.0f * x) + 1.0f); }

__device__ __forceinline__ f32x8 wmma_bf16(bf16x16 a, bf16x16 b, f32x8 c) {
    return __builtin_amdgcn_wmma_f32_16x16x32_bf16(false, a, false, b, (short)0, c, false, false);
}

// A-operand k-swizzle: within each 32-wide k chunk store order is
// contiguous [k0..31] (identity, matches B's contiguous k order).
// Lane l<16 (row l) reads bytes [kc*64 + 0, +32); lane l+16 reads [+32, +64).
__device__ __forceinline__ int kswiz(int k) {
    return k;
}

__device__ __forceinline__ bf16x16 load_a_lds(const __bf16* rowbase, int lane, int kc) {
    return *(const bf16x16*)(rowbase + kc * 32 + ((lane & 16) ? 16 : 0));
}

// B operand (32x16 bf16, B[k][n] = W[n][k]): lane n<16 -> k 0..15 of chunk,
// lane n+16 -> k 16..31. One contiguous 32B read of W's row n.
__device__ __forceinline__ bf16x16 load_b_glb(const __bf16* W, int ncol, int lane, int kc) {
    const __bf16* p = W + (size_t)ncol * DKn + kc * 32 + ((lane & 16) ? 16 : 0);
    return *(const bf16x16*)p;
}

// ------------------------------- precompute --------------------------------

__global__ void k_zero_col0(float* pred) { pred[threadIdx.x * Sn] = 0.0f; }

__global__ void k_w1c(const float* W1, float* w1c) {
    int d = threadIdx.x;
    float a = 0.f;
    for (int j = 0; j < 128; ++j) a += W1[d * 640 + 512 + j];
    w1c[d] = a;
}

// M[e,k] = sum_d W[e, woff+d] * W1[d, w1off+k]
__global__ __launch_bounds__(256) void k_comp(const float* W, int ldw, int woff,
                                              const float* W1, int w1off, float* M) {
    int e = blockIdx.x, k = threadIdx.x;
    __shared__ float Wr[256];
    Wr[k] = W[e * ldw + woff + k];
    __syncthreads();
    float acc = 0.f;
#pragma unroll 4
    for (int d = 0; d < 256; ++d) acc += Wr[d] * W1[d * 640 + w1off + k];
    M[e * 256 + k] = acc;
}

// v[e] = sum_d W[e, woff+d] * u[d]
__global__ void k_vecdot(const float* W, int ldw, int woff, const float* u, float* v) {
    int e = threadIdx.x;
    float acc = 0.f;
    for (int d = 0; d < 256; ++d) acc += W[e * ldw + woff + d] * u[d];
    v[e] = acc;
}

// T[(n*256+e)*ostr + ooff] = sum_k E[n,k]*M[e*ldm+moff+k] + add1[e] + add2[e]
__global__ __launch_bounds__(256) void k_table(const float* E, const float* M, int ldm, int moff,
                                               const float* add1, const float* add2,
                                               float* T, int ostr, int ooff) {
    int n = blockIdx.x, e = threadIdx.x;
    __shared__ float Es[256];
    Es[e] = E[n * 256 + e];
    __syncthreads();
    const float* Mr = M + (size_t)e * ldm + moff;
    float acc = 0.f;
#pragma unroll 8
    for (int k = 0; k < 256; ++k) acc += Es[k] * Mr[k];
    if (add1) acc += add1[e];
    if (add2) acc += add2[e];
    T[((size_t)n * 256 + e) * ostr + ooff] = acc;
}

// per-question scalars: qds[n] = sum_e sigmoid(E_q[n]·Wd[e]+bd[e])*qm[n,e]
//                       discq[n] = 10*sigmoid(E_q[n]·Wdc+bdc)
__global__ __launch_bounds__(256) void k_qscalars(const float* E_q, const float* Wd, const float* bd,
                                                  const float* Wdc, const float* bdc, const float* qm,
                                                  float* discq, float* qds) {
    int n = blockIdx.x, e = threadIdx.x;
    __shared__ float Es[256];
    __shared__ float red[256];
    Es[e] = E_q[n * 256 + e];
    __syncthreads();
    float acc = bd[e];
    for (int k = 0; k < 256; ++k) acc += Es[k] * Wd[e * 256 + k];
    red[e] = sigm(acc) * qm[(size_t)n * 256 + e];
    __syncthreads();
    for (int s = 128; s > 0; s >>= 1) { if (e < s) red[e] += red[e + s]; __syncthreads(); }
    if (e == 0) qds[n] = red[0];
    __syncthreads();
    red[e] = Es[e] * Wdc[e];
    __syncthreads();
    for (int s = 128; s > 0; s >>= 1) { if (e < s) red[e] += red[e + s]; __syncthreads(); }
    if (e == 0) discq[n] = 10.0f * sigm(red[0] + bdc[0]);
}

__global__ void k_cvt_bf16(const float* src, int lds, int off, __bf16* dst) {
    int i = blockIdx.x * 256 + threadIdx.x;          // 65536 elements
    int e = i >> 8, k = i & 255;
    dst[i] = (__bf16)src[e * lds + off + k];
}

// ------------------------------- recurrence --------------------------------

struct RecArgs {
    const int   *q, *it, *ut;
    const float *corr, *h0, *ba, *qmat;
    const float2 *TqA, *TutA, *TqC, *TutC, *TitB;   // packed (W2-term, W3-term)
    const float *v2a, *v2c, *v3a, *v3c;
    const float *Tit4, *discq, *qds;
    const __bf16 *W2d, *W3d, *W4a, *W4b, *Wab;
    float *pred;
};

#define H32S 260    // fp32 row stride (bank shift 4)
#define HBFA 272    // bf16 A-swizzled row stride (544B: 32B-aligned rows)

__global__ __launch_bounds__(256, 1) void k_recurrent(RecArgs A) {
    __shared__ float  h32[2][16][H32S];
    __shared__ alignas(32) __bf16 hbf[2][16][HBFA];   // A-operand k-swizzled
    __shared__ alignas(32) __bf16 lgbf[16][HBFA];     // A-operand k-swizzled
    __shared__ unsigned oqp[16], outp[16], oit[16], oqc[16], outc[16], oqn[16];
    __shared__ float cprev[16], ccur[16], disc_s[16], qds_s[16], ysum[16];

    const int tid  = threadIdx.x;
    const int lane = tid & 31;
    const int w    = tid >> 5;
    const int half = (lane >> 4) & 1;
    const int ln   = lane & 15;
    const int b0   = blockIdx.x * 16;

    // init state from h0
    for (int i = tid; i < 16 * DKn; i += 256) {
        int r = i >> 8, c = i & 255;
        float v = A.h0[(size_t)(b0 + r) * DKn + c];
        h32[0][r][c] = v;
        hbf[0][r][kswiz(c)] = (__bf16)v;
    }
    if (tid < 16) ysum[tid] = 0.f;
    __syncthreads();

    int cur = 0;
    for (int t = 0; t < TSn; ++t) {
        const int nxt = cur ^ 1;
        // ---- phase 0: per-row pre-scaled 32-bit gather offsets / scalars ----
        if (tid < 16) {
            int b = b0 + tid;
            oit[tid]  = (unsigned)A.it[b * Sn + t] * DKn;
            oqc[tid]  = (unsigned)A.q [b * Sn + t] * DKn;
            outc[tid] = (unsigned)A.ut[b * Sn + t] * DKn;
            ccur[tid] = A.corr[b * Sn + t];
            if (t > 0) {
                oqp[tid]  = (unsigned)A.q [b * Sn + t - 1] * DKn;
                outp[tid] = (unsigned)A.ut[b * Sn + t - 1] * DKn;
                cprev[tid]= A.corr[b * Sn + t - 1];
            }
            int qn = A.q[b * Sn + t + 1];
            oqn[tid] = (unsigned)qn * DKn;
            disc_s[tid] = A.discq[qn];
            qds_s[tid]  = A.qds[qn];
            ysum[tid] = 0.f;
        }
        __syncthreads();

        const bool has_prev = (t > 0);
        float lgreg[2][8];

        // ---- phase 1: lg = tanh(pre2 + h@W2d^T), gl = sigm(pre3 + h@W3d^T) ----
#pragma unroll
        for (int ti = 0; ti < 2; ++ti) {
            const int n  = (w * 2 + ti) * 16 + ln;
            const int sn = kswiz(n);
            const float v2cn = A.v2c[n], v3cn = A.v3c[n];
            f32x8 acc2, acc3;
#pragma unroll
            for (int j = 0; j < 8; ++j) {
                int m = j + 8 * half;
                float2 pb = A.TitB[oit[m]  + n];
                float2 pc = A.TqC [oqc[m]  + n];
                float2 pu = A.TutC[outc[m] + n];
                acc2[j] = pb.x + pc.x + pu.x + ccur[m] * v2cn;
                acc3[j] = pb.y + pc.y + pu.y + ccur[m] * v3cn;
            }
            if (has_prev) {
                const float v2an = A.v2a[n], v3an = A.v3a[n];
#pragma unroll
                for (int j = 0; j < 8; ++j) {
                    int m = j + 8 * half;
                    float2 pa = A.TqA [oqp[m]  + n];
                    float2 pu = A.TutA[outp[m] + n];
                    acc2[j] += pa.x + pu.x + cprev[m] * v2an;
                    acc3[j] += pa.y + pu.y + cprev[m] * v3an;
                }
            }
#pragma unroll
            for (int kc = 0; kc < 8; ++kc) {
                bf16x16 av = load_a_lds(&hbf[cur][ln][0], lane, kc);
                bf16x16 b2 = load_b_glb(A.W2d, n, lane, kc);
                bf16x16 b3 = load_b_glb(A.W3d, n, lane, kc);
                acc2 = wmma_bf16(av, b2, acc2);
                acc3 = wmma_bf16(av, b3, acc3);
            }
#pragma unroll
            for (int j = 0; j < 8; ++j) {
                int m = j + 8 * half;
                float lg = tanh_fast(acc2[j]);
                float gl = sigm(acc3[j]);
                float LG = gl * (lg + 1.0f) * 0.5f;
                lgreg[ti][j] = LG;
                lgbf[m][sn] = (__bf16)LG;
            }
        }
        __syncthreads();

        // ---- phase 2: gf = sigm(pre4 + h@W4a^T + LG@W4b^T); h' = LG + gf*h ----
#pragma unroll
        for (int ti = 0; ti < 2; ++ti) {
            const int n  = (w * 2 + ti) * 16 + ln;
            const int sn = kswiz(n);
            f32x8 acc4;
#pragma unroll
            for (int j = 0; j < 8; ++j) {
                int m = j + 8 * half;
                acc4[j] = A.Tit4[oit[m] + n];
            }
#pragma unroll
            for (int kc = 0; kc < 8; ++kc) {
                bf16x16 ah  = load_a_lds(&hbf[cur][ln][0], lane, kc);
                bf16x16 b4a = load_b_glb(A.W4a, n, lane, kc);
                acc4 = wmma_bf16(ah, b4a, acc4);
                bf16x16 al  = load_a_lds(&lgbf[ln][0], lane, kc);
                bf16x16 b4b = load_b_glb(A.W4b, n, lane, kc);
                acc4 = wmma_bf16(al, b4b, acc4);
            }
#pragma unroll
            for (int j = 0; j < 8; ++j) {
                int m = j + 8 * half;
                float gf = sigm(acc4[j]);
                float hn = lgreg[ti][j] + gf * h32[cur][m][n];
                h32[nxt][m][n] = hn;
                hbf[nxt][m][sn] = (__bf16)hn;
            }
        }
        __syncthreads();

        // ---- phase 3: ability = sigm(h'@Wa^T + ba); y reduction ----
#pragma unroll
        for (int ti = 0; ti < 2; ++ti) {
            const int n  = (w * 2 + ti) * 16 + ln;
            const float ban = A.ba[n];
            f32x8 acc5;
#pragma unroll
            for (int j = 0; j < 8; ++j) acc5[j] = ban;
#pragma unroll
            for (int kc = 0; kc < 8; ++kc) {
                bf16x16 ah  = load_a_lds(&hbf[nxt][ln][0], lane, kc);
                bf16x16 ba_ = load_b_glb(A.Wab, n, lane, kc);
                acc5 = wmma_bf16(ah, ba_, acc5);
            }
#pragma unroll
            for (int j = 0; j < 8; ++j) {
                int m = j + 8 * half;
                float v = sigm(acc5[j]) * A.qmat[oqn[m] + n];
#pragma unroll
                for (int off = 8; off >= 1; off >>= 1) v += __shfl_xor(v, off, 16);
                if (ln == 0) atomicAdd(&ysum[m], v);
            }
        }
        __syncthreads();

        if (tid < 16) {
            float s = disc_s[tid] * (ysum[tid] - qds_s[tid]);
            A.pred[(size_t)(b0 + tid) * Sn + t + 1] = sigm(s);
        }
        cur = nxt;
        __syncthreads();
    }
}

// --------------------------------- launch ----------------------------------

extern "C" void kernel_launch(void* const* d_in, const int* in_sizes, int n_in,
                              void* d_out, int out_size, void* d_ws, size_t ws_size,
                              hipStream_t stream) {
    const int*   qseq = (const int*)d_in[0];
    const int*   itq  = (const int*)d_in[1];
    const int*   utq  = (const int*)d_in[2];
    const float* corr = (const float*)d_in[3];
    const float* h0   = (const float*)d_in[4];
    const float* E_q  = (const float*)d_in[5];
    const float* E_it = (const float*)d_in[6];
    const float* E_ut = (const float*)d_in[7];
    const float* W1   = (const float*)d_in[8];
    const float* b1   = (const float*)d_in[9];
    const float* W2   = (const float*)d_in[10];
    const float* b2   = (const float*)d_in[11];
    const float* W3   = (const float*)d_in[12];
    const float* b3   = (const float*)d_in[13];
    const float* W4   = (const float*)d_in[14];
    const float* b4   = (const float*)d_in[15];
    const float* Wa   = (const float*)d_in[16];
    const float* ba   = (const float*)d_in[17];
    const float* Wd   = (const float*)d_in[18];
    const float* bd   = (const float*)d_in[19];
    const float* Wdc  = (const float*)d_in[20];
    const float* bdc  = (const float*)d_in[21];
    const float* qmat = (const float*)d_in[22];
    float* pred = (float*)d_out;

    // ---- workspace carve ----
    float* F = (float*)d_ws;
    size_t o = 0;
    auto A = [&](size_t n) { float* r = F + o; o += n; return r; };
    float* w1c  = A(256);
    float* M2a  = A(65536); float* M2au = A(65536); float* M2c = A(65536); float* M2cu = A(65536);
    float* M3a  = A(65536); float* M3au = A(65536); float* M3c = A(65536); float* M3cu = A(65536);
    float* v2a  = A(256);   float* v2c  = A(256);   float* v3a = A(256);   float* v3c  = A(256);
    float* k2a  = A(256);   float* k2c  = A(256);   float* k3a = A(256);   float* k3c  = A(256);
    float* TqA  = A((size_t)NQ1 * 512);   // packed {Tq2a, Tq3a}
    float* TqC  = A((size_t)NQ1 * 512);   // packed {Tq2c, Tq3c}
    float* TutA = A((size_t)NT1 * 512);   // packed {Tut2a, Tut3a}
    float* TutC = A((size_t)NT1 * 512);   // packed {Tut2c, Tut3c}
    float* TitB = A((size_t)NT1 * 512);   // packed {Tit2b, Tit3b}
    float* Tit4 = A((size_t)NT1 * 256);
    float* discq = A(NQ1); float* qds = A(NQ1);
    __bf16* BF  = (__bf16*)(F + o);
    __bf16* W2d = BF;            __bf16* W3d = BF + 65536;
    __bf16* W4a = BF + 131072;   __bf16* W4b = BF + 196608;
    __bf16* Wab = BF + 262144;

    k_zero_col0<<<1, 256, 0, stream>>>(pred);
    k_w1c<<<1, 256, 0, stream>>>(W1, w1c);

    // composites: block(a)=l_pre, block(c)=learning; W1q cols 0, W1ut cols 256
    k_comp<<<256, 256, 0, stream>>>(W2, 1024,   0, W1,   0, M2a);
    k_comp<<<256, 256, 0, stream>>>(W2, 1024,   0, W1, 256, M2au);
    k_comp<<<256, 256, 0, stream>>>(W2, 1024, 512, W1,   0, M2c);
    k_comp<<<256, 256, 0, stream>>>(W2, 1024, 512, W1, 256, M2cu);
    k_comp<<<256, 256, 0, stream>>>(W3, 1024,   0, W1,   0, M3a);
    k_comp<<<256, 256, 0, stream>>>(W3, 1024,   0, W1, 256, M3au);
    k_comp<<<256, 256, 0, stream>>>(W3, 1024, 512, W1,   0, M3c);
    k_comp<<<256, 256, 0, stream>>>(W3, 1024, 512, W1, 256, M3cu);

    k_vecdot<<<1, 256, 0, stream>>>(W2, 1024,   0, w1c, v2a);
    k_vecdot<<<1, 256, 0, stream>>>(W2, 1024, 512, w1c, v2c);
    k_vecdot<<<1, 256, 0, stream>>>(W3, 1024,   0, w1c, v3a);
    k_vecdot<<<1, 256, 0, stream>>>(W3, 1024, 512, w1c, v3c);
    k_vecdot<<<1, 256, 0, stream>>>(W2, 1024,   0, b1, k2a);
    k_vecdot<<<1, 256, 0, stream>>>(W2, 1024, 512, b1, k2c);
    k_vecdot<<<1, 256, 0, stream>>>(W3, 1024,   0, b1, k3a);
    k_vecdot<<<1, 256, 0, stream>>>(W3, 1024, 512, b1, k3c);

    // packed tables (x = W2 term, y = W3 term); constants folded:
    // prev-block table gets W@b1; cur it-block table gets W@b1 + bias
    k_table<<<NQ1, 256, 0, stream>>>(E_q,  M2a,  256,   0, k2a, nullptr, TqA, 2, 0);
    k_table<<<NQ1, 256, 0, stream>>>(E_q,  M3a,  256,   0, k3a, nullptr, TqA, 2, 1);
    k_table<<<NQ1, 256, 0, stream>>>(E_q,  M2c,  256,   0, nullptr, nullptr, TqC, 2, 0);
    k_table<<<NQ1, 256, 0, stream>>>(E_q,  M3c,  256,   0, nullptr, nullptr, TqC, 2, 1);
    k_table<<<NT1, 256, 0, stream>>>(E_ut, M2au, 256,   0, nullptr, nullptr, TutA, 2, 0);
    k_table<<<NT1, 256, 0, stream>>>(E_ut, M3au, 256,   0, nullptr, nullptr, TutA, 2, 1);
    k_table<<<NT1, 256, 0, stream>>>(E_ut, M2cu, 256,   0, nullptr, nullptr, TutC, 2, 0);
    k_table<<<NT1, 256, 0, stream>>>(E_ut, M3cu, 256,   0, nullptr, nullptr, TutC, 2, 1);
    k_table<<<NT1, 256, 0, stream>>>(E_it, W2,  1024, 256, k2c, b2, TitB, 2, 0);
    k_table<<<NT1, 256, 0, stream>>>(E_it, W3,  1024, 256, k3c, b3, TitB, 2, 1);
    k_table<<<NT1, 256, 0, stream>>>(E_it, W4,   768, 512, b4, nullptr, Tit4, 1, 0);

    k_qscalars<<<NQ1, 256, 0, stream>>>(E_q, Wd, bd, Wdc, bdc, qmat, discq, qds);

    k_cvt_bf16<<<256, 256, 0, stream>>>(W2, 1024, 768, W2d);
    k_cvt_bf16<<<256, 256, 0, stream>>>(W3, 1024, 768, W3d);
    k_cvt_bf16<<<256, 256, 0, stream>>>(W4,  768,   0, W4a);
    k_cvt_bf16<<<256, 256, 0, stream>>>(W4,  768, 256, W4b);
    k_cvt_bf16<<<256, 256, 0, stream>>>(Wa,  256,   0, Wab);

    RecArgs ra;
    ra.q = qseq; ra.it = itq; ra.ut = utq; ra.corr = corr; ra.h0 = h0; ra.ba = ba; ra.qmat = qmat;
    ra.TqA = (const float2*)TqA; ra.TutA = (const float2*)TutA;
    ra.TqC = (const float2*)TqC; ra.TutC = (const float2*)TutC;
    ra.TitB = (const float2*)TitB;
    ra.v2a = v2a; ra.v2c = v2c; ra.v3a = v3a; ra.v3c = v3c;
    ra.Tit4 = Tit4; ra.discq = discq; ra.qds = qds;
    ra.W2d = W2d; ra.W3d = W3d; ra.W4a = W4a; ra.W4b = W4b; ra.Wab = Wab;
    ra.pred = pred;

    k_recurrent<<<Bn / 16, 256, 0, stream>>>(ra);
}
